// InstantNGP_26010321945203
// MI455X (gfx1250) — matrix operand order repaired and checked by reading, not compiled, not run on hardware
//
#include <hip/hip_runtime.h>

typedef __attribute__((ext_vector_type(16))) _Float16 v16h;
typedef __attribute__((ext_vector_type(8)))  float    v8f;
typedef __attribute__((ext_vector_type(4)))  float    v4f;

namespace {
constexpr int   kT        = 1 << 19;       // hash table entries per level
constexpr int   kS        = 1024;
constexpr int   kWaves    = 8;             // waves per block (wave32)
constexpr int   kPtsBlk   = 16 * kWaves;   // 128 points per block
constexpr float kEncScale = 8192.0f;       // 2^13: lift +-1e-4 feats into f16 normal range
constexpr float kInvScale = 1.0f / 8192.0f;

// LDS byte offsets
constexpr int kW0T  = 0;      // W0^T [64][32] f16 -> 4096 B
constexpr int kW1T  = 4096;   // W1^T [64][64] f16 -> 8192 B
constexpr int kW2T  = 12288;  // W2^T [16][64] f16 -> 2048 B (rows n>=3 zero)
constexpr int kWsc  = 14336;  // per-wave: enc [16][32] f16 (1024B) + h [16][64] f16 (2048B)
constexpr int kWstr = 3072;
constexpr int kLds  = kWsc + kWaves * kWstr; // 38912 B
}

// A-fragment, 16-bit A 16x32 layout: lanes 0-15 rows M=0..15 hold K k0..k0+7 (v0-3)
// and k0+16..k0+23 (v4-7) with k0 = (lane>=16)*8.
__device__ __forceinline__ v16h load_a_frag(const _Float16* base, int rowStride,
                                            int kOff, int lane) {
  union { v16h h; v4f f[2]; } u;
  const int r  = lane & 15;
  const int k0 = ((lane >> 4) << 3) + kOff;
  u.f[0] = *(const v4f*)(base + r * rowStride + k0);
  u.f[1] = *(const v4f*)(base + r * rowStride + k0 + 16);
  return u.h;
}

// B-fragment, 16-bit B 32x16 layout: lane n=lane&15 holds column n, K k0..k0+7 (v0-3)
// and k0+16..k0+23 (v4-7) with k0 = (lane>=16)*8. base is the transposed weight (row = N).
__device__ __forceinline__ v16h load_b_frag(const _Float16* base, int rowStride,
                                            int nOff, int kOff, int lane) {
  union { v16h h; v4f f[2]; } u;
  const int n  = (lane & 15) + nOff;
  const int k0 = ((lane >> 4) << 3) + kOff;
  u.f[0] = *(const v4f*)(base + n * rowStride + k0);
  u.f[1] = *(const v4f*)(base + n * rowStride + k0 + 16);
  return u.h;
}

__global__ __launch_bounds__(256) void ngp_fused_kernel(
    const float* __restrict__ xy, const float* __restrict__ ht,
    const float* __restrict__ W0, const float* __restrict__ W1,
    const float* __restrict__ W2, float* __restrict__ out) {
  __shared__ __align__(16) unsigned char smem[kLds];
  _Float16* w0t = (_Float16*)(smem + kW0T);
  _Float16* w1t = (_Float16*)(smem + kW1T);
  _Float16* w2t = (_Float16*)(smem + kW2T);

  const int tid = threadIdx.x;

  // ---- cooperative weight load: f32 -> f16, transposed (row = output unit N) ----
  for (int i = tid; i < 32 * 64; i += 256) {        // W0 (32 x 64)
    const int k = i >> 6, n = i & 63;
    w0t[n * 32 + k] = (_Float16)W0[i];
  }
  for (int i = tid; i < 64 * 64; i += 256) {        // W1 (64 x 64)
    const int k = i >> 6, n = i & 63;
    w1t[n * 64 + k] = (_Float16)W1[i];
  }
  for (int i = tid; i < 16 * 64; i += 256) {        // W2 (64 x 3), N padded to 16
    const int n = i >> 6, k = i & 63;
    w2t[i] = (n < 3) ? (_Float16)W2[k * 3 + n] : (_Float16)0.0f;
  }
  __syncthreads();

  const int wave = tid >> 5;
  const int lane = tid & 31;
  const int hi   = lane >> 4;                       // lane half-group
  _Float16* encB = (_Float16*)(smem + kWsc + wave * kWstr);          // [16][32]
  _Float16* hB   = (_Float16*)(smem + kWsc + wave * kWstr + 1024);   // [16][64]

  const int pointBase = blockIdx.x * kPtsBlk + wave * 16;
  const int gp = pointBase + (lane & 15);
  const float x0 = xy[2 * gp + 0];
  const float x1 = xy[2 * gp + 1];

  // floor(16 * 1.5^l) for l = 0..15 (exact in fp32)
  constexpr int RES[16] = {16, 24, 36, 54, 81, 121, 182, 273,
                           410, 615, 922, 1383, 2075, 3113, 4670, 7006};

  // ---- hash-grid encoding: lanes 0-15 even levels, 16-31 odd levels ----
#pragma unroll
  for (int t = 0; t < 8; ++t) {
    const int lev = 2 * t + hi;
    const int res = hi ? RES[2 * t + 1] : RES[2 * t];   // compile-time constants
    const bool dense = ((res + 1) * (res + 1) <= kT);
    const float rf = (float)res;
    const float* tab = ht + (size_t)lev * (size_t)(kT * 2);

    const float p0f = x0 * rf, p1f = x1 * rf;
    const int   i0 = (int)floorf(p0f), i1 = (int)floorf(p1f);
    const float w0 = p0f - (float)i0,  w1 = p1f - (float)i1;

    float a0 = 0.f, a1 = 0.f;
#pragma unroll
    for (int c = 0; c < 4; ++c) {               // corners [0,0],[0,1],[1,0],[1,1]
      const int ca = i0 + (c >> 1);
      const int cb = i1 + (c & 1);
      const unsigned hidx =
          (((unsigned)ca) ^ (((unsigned)cb) * 2654435761u)) & (unsigned)(kT - 1);
      const int didx = ca + cb * (res + 1);
      const int idx = dense ? didx : (int)hidx;
      const float f0 = tab[2 * idx + 0];
      const float f1 = tab[2 * idx + 1];
      const float wt = ((c >> 1) ? w0 : 1.f - w0) * ((c & 1) ? w1 : 1.f - w1);
      a0 = fmaf(f0, wt, a0);
      a1 = fmaf(f1, wt, a1);
    }
    const int p = lane & 15;
    encB[p * 32 + lev * 2 + 0] = (_Float16)(a0 * kEncScale);
    encB[p * 32 + lev * 2 + 1] = (_Float16)(a1 * kEncScale);
  }
  __syncthreads();   // also fences LDS before fragment loads

  // ---- layer 0: (16x32) @ (32x64), K = 32 = one WMMA per N-tile ----
  {
    const v16h a = load_a_frag(encB, 32, 0, lane);
    v8f acc[4];
#pragma unroll
    for (int nt = 0; nt < 4; ++nt) {
      const v16h b = load_b_frag(w0t, 32, nt * 16, 0, lane);
      v8f z = {};
      acc[nt] = __builtin_amdgcn_wmma_f32_16x16x32_f16(false, a, false, b,
                                                       (short)0, z, false, false);
    }
    // ReLU (scale-invariant: keep 2^13 factor) -> h in LDS, row = point
#pragma unroll
    for (int nt = 0; nt < 4; ++nt) {
      const int n = (lane & 15) + nt * 16;
#pragma unroll
      for (int v = 0; v < 8; ++v) {
        const float x = fmaxf(acc[nt][v], 0.0f);    // single v_max_num_f32
        hB[(v + hi * 8) * 64 + n] = (_Float16)x;
      }
    }
  }

  // ---- layer 1: (16x64) @ (64x64), K split 0..31 / 32..63 ----
  {
    const v16h a0 = load_a_frag(hB, 64, 0, lane);
    const v16h a1 = load_a_frag(hB, 64, 32, lane);
    v8f acc[4];
#pragma unroll
    for (int nt = 0; nt < 4; ++nt) {
      const v16h b0 = load_b_frag(w1t, 64, nt * 16, 0, lane);
      v8f z = {};
      v8f c = __builtin_amdgcn_wmma_f32_16x16x32_f16(false, a0, false, b0,
                                                     (short)0, z, false, false);
      const v16h b1 = load_b_frag(w1t, 64, nt * 16, 32, lane);
      acc[nt] = __builtin_amdgcn_wmma_f32_16x16x32_f16(false, a1, false, b1,
                                                       (short)0, c, false, false);
    }
#pragma unroll
    for (int nt = 0; nt < 4; ++nt) {
      const int n = (lane & 15) + nt * 16;
#pragma unroll
      for (int v = 0; v < 8; ++v) {
        const float x = fmaxf(acc[nt][v], 0.0f);
        hB[(v + hi * 8) * 64 + n] = (_Float16)x;
      }
    }
  }

  // ---- layer 2: (16x64) @ (64x3), N padded to one 16-tile ----
  {
    const v16h a0 = load_a_frag(hB, 64, 0, lane);
    const v16h a1 = load_a_frag(hB, 64, 32, lane);
    const v16h b0 = load_b_frag(w2t, 64, 0, 0, lane);
    const v16h b1 = load_b_frag(w2t, 64, 0, 32, lane);
    v8f z = {};
    v8f o = __builtin_amdgcn_wmma_f32_16x16x32_f16(false, a0, false, b0,
                                                   (short)0, z, false, false);
    o = __builtin_amdgcn_wmma_f32_16x16x32_f16(false, a1, false, b1,
                                               (short)0, o, false, false);
    // out layout (3, S, S): out[c*S*S + point]; undo the 2^13 encoding scale
    const int n = lane & 15;
    if (n < 3) {
      float* op = out + (size_t)n * (size_t)(kS * kS) + pointBase + hi * 8;
#pragma unroll
      for (int v = 0; v < 8; ++v) op[v] = o[v] * kInvScale;
    }
  }
}

extern "C" void kernel_launch(void* const* d_in, const int* in_sizes, int n_in,
                              void* d_out, int out_size, void* d_ws, size_t ws_size,
                              hipStream_t stream) {
  const float* xy = (const float*)d_in[0];   // (S*S, 2)
  const float* ht = (const float*)d_in[1];   // (L, T, F)
  const float* W0 = (const float*)d_in[2];   // (32, 64)
  const float* W1 = (const float*)d_in[3];   // (64, 64)
  const float* W2 = (const float*)d_in[4];   // (64, 3)
  float* out = (float*)d_out;                // (3, S, S)

  const int nPoints = kS * kS;               // 1048576
  const int blocks = nPoints / kPtsBlk;      // 8192
  ngp_fused_kernel<<<blocks, 256, 0, stream>>>(xy, ht, W0, W1, W2, out);
}
